// SwinTransformerBlock_25692494364868
// MI455X (gfx1250) — compile-verified, matrix operand order checked by reading, not run on hardware
//
#include <hip/hip_runtime.h>
#include <hip/hip_bf16.h>
#include <math.h>

// ---------------- constants ----------------
#define IMGH 256
#define IMGW 256
#define CDIM 384
#define NHEADS 12
#define HDIM 32
#define WSZ 8
#define NTOK 64          // tokens per window
#define SSH 4
#define HID 1536
#define NWIN 2048        // total windows (B=2, 32x32 windows per image)
#define LOG100F 4.605170185988092f

typedef __attribute__((ext_vector_type(16))) __bf16 v16bf;
typedef __attribute__((ext_vector_type(8)))  float v8f;
typedef __attribute__((ext_vector_type(4)))  int   v4i;

union FragA { v16bf bf; unsigned short u[16]; };

// ---- optional CDNA5 async global->LDS path (guarded; falls back cleanly) ----
#if defined(__AMDGCN__) && __has_builtin(__builtin_amdgcn_global_load_async_to_lds_b128) && \
    __has_builtin(__builtin_amdgcn_s_wait_asynccnt)
#define HAVE_ASYNC_LDS 1
#else
#define HAVE_ASYNC_LDS 0
#endif

#if HAVE_ASYNC_LDS
typedef __attribute__((address_space(1))) v4i g_v4i;   // global int4
typedef __attribute__((address_space(3))) v4i l_v4i;   // LDS int4
__device__ __forceinline__ void async_g2l_16B(void* l, const void* g) {
    __builtin_amdgcn_global_load_async_to_lds_b128((g_v4i*)g, (l_v4i*)l, 0, 0);
}
__device__ __forceinline__ void async_wait0() { __builtin_amdgcn_s_wait_asynccnt(0); }
#endif

// fp32 -> bf16 (round-to-nearest-even)
__device__ __forceinline__ unsigned short f2bf(float f) {
    unsigned int u = __float_as_uint(f);
    unsigned int r = u + 0x7FFFu + ((u >> 16) & 1u);
    return (unsigned short)(r >> 16);
}
__device__ __forceinline__ float bf2f(unsigned short h) {
    return __uint_as_float(((unsigned int)h) << 16);
}

// ISA 16-bit A/B fragment K index for element j of the 16-half fragment.
// lane (0..15) holds K in {0..7, 16..23}; lane+16 holds {8..15, 24..31}.
__device__ __forceinline__ int kmap(int j, int hf) {
    return (j & 7) + (hf << 3) + ((j & 8) << 1);
}

__device__ __forceinline__ v16bf load_frag(const unsigned short* p, int hf) {
    FragA f;
#pragma unroll
    for (int j = 0; j < 16; ++j) f.u[j] = p[kmap(j, hf)];
    return f.bf;
}
__device__ __forceinline__ v16bf load_frag_strided(const unsigned short* p, int stride, int hf) {
    FragA f;
#pragma unroll
    for (int j = 0; j < 16; ++j) f.u[j] = p[kmap(j, hf) * stride];
    return f.bf;
}

__device__ __forceinline__ v8f wmma_bf16(v16bf a, v16bf b, v8f c) {
    return __builtin_amdgcn_wmma_f32_16x16x32_bf16(false, a, false, b, (short)0, c, false, false);
}

__device__ __forceinline__ float gelu_exact(float v) {
    return 0.5f * v * (1.0f + erff(v * 0.70710678118654752f));
}

// ---------------- prep kernels ----------------
__global__ void cvt_bf16_kernel(const float* __restrict__ s, unsigned short* __restrict__ d, int n) {
    int i = blockIdx.x * blockDim.x + threadIdx.x;
    if (i < n) d[i] = f2bf(s[i]);
}

__device__ __forceinline__ float coordval(int i) {
    float c = (float)(i - 7) * (8.0f / 7.0f);
    float sgn = (c > 0.f) ? 1.f : ((c < 0.f) ? -1.f : 0.f);
    return sgn * log2f(fabsf(c) + 1.f) * (1.0f / 3.0f);  // /log2(8)
}

__global__ __launch_bounds__(256)
void cpb_kernel(const float* __restrict__ w1, const float* __restrict__ b1,
                const float* __restrict__ w2, const float* __restrict__ ls,
                float* __restrict__ scale, float* __restrict__ rpb) {
    __shared__ float btab[225 * NHEADS];
    int tid = threadIdx.x;
    if (tid < NHEADS) scale[tid] = expf(fminf(ls[tid], LOG100F));
    for (int e = tid; e < 225; e += 256) {
        float v0 = coordval(e / 15);
        float v1 = coordval(e % 15);
        float o[NHEADS];
#pragma unroll
        for (int hh = 0; hh < NHEADS; ++hh) o[hh] = 0.f;
        for (int k = 0; k < 512; ++k) {
            float hv = fmaxf(w1[k * 2] * v0 + w1[k * 2 + 1] * v1 + b1[k], 0.f);
#pragma unroll
            for (int hh = 0; hh < NHEADS; ++hh) o[hh] += w2[hh * 512 + k] * hv;
        }
#pragma unroll
        for (int hh = 0; hh < NHEADS; ++hh) btab[e * NHEADS + hh] = o[hh];
    }
    __syncthreads();
    for (int z = tid; z < NHEADS * NTOK * NTOK; z += 256) {
        int h = z >> 12; int rem = z & 4095;
        int i = rem >> 6, j = rem & 63;
        int r1 = i >> 3, c1 = i & 7, r2 = j >> 3, c2 = j & 7;
        int idx = (r1 - r2 + 7) * 15 + (c1 - c2 + 7);
        float v = btab[idx * NHEADS + h];
        rpb[z] = 16.f / (1.f + expf(-v));
    }
}

// ---------------- fused attention kernel (one window per block) ----------------
// shared-memory layout (bytes)
#define SM_XW   0                      // attn-in / attn-out: ushort [64][392] = 50176
#define SM_QKV  50432                  // qkv ushort [64][1160] = 148480 ; also f32 staging / proj f32
#define SM_SC   (SM_QKV + 148480)      // scores f32 [64][68] = 17408
#define SM_QN   (SM_SC + 17408)        // qn ushort [64][36] = 4608 ; qn+kn reused as P ushort [64][68]
#define SM_KN   (SM_QN + 4608)
#define SM_MISC (SM_KN + 4608)         // gidx[64] int + rid[64] int
#define SMEM_ATTN (SM_MISC + 512)

__global__ __launch_bounds__(256)
void swin_attn_kernel(const float* __restrict__ x,
                      const unsigned short* __restrict__ qkvw,
                      const float* __restrict__ qb, const float* __restrict__ vb,
                      const float* __restrict__ scale, const float* __restrict__ rpb,
                      const unsigned short* __restrict__ projw, const float* __restrict__ projb,
                      const float* __restrict__ g1, const float* __restrict__ b1,
                      float* __restrict__ x1out) {
    extern __shared__ char smem[];
    unsigned short* xw   = (unsigned short*)(smem + SM_XW);    // stride 392
    unsigned short* qkv  = (unsigned short*)(smem + SM_QKV);   // stride 1160
    float*          sc   = (float*)(smem + SM_SC);             // stride 68
    unsigned short* qn   = (unsigned short*)(smem + SM_QN);    // stride 36
    unsigned short* kn   = (unsigned short*)(smem + SM_KN);    // stride 36
    unsigned short* pbuf = qn;                                  // stride 68 (spans qn+kn)
    int* gidx = (int*)(smem + SM_MISC);
    int* rid  = gidx + 64;

    const int tid = threadIdx.x;
    const int wave = tid >> 5, lane = tid & 31;
    const int hf = lane >> 4, ln = lane & 15;

    const int wid = blockIdx.x;
    const int b = wid >> 10;
    const int wrem = wid & 1023;
    const int wh = wrem >> 5, wwi = wrem & 31;

    // phase 0: token -> source-coord gather index & shift-mask region id
    if (tid < NTOK) {
        int r = tid >> 3, c0 = tid & 7;
        int hsf = wh * WSZ + r;          // shifted-frame coords
        int wsf = wwi * WSZ + c0;
        int sh = (hsf + SSH) & (IMGH - 1);
        int sw = (wsf + SSH) & (IMGW - 1);
        gidx[tid] = (b << 16) + sh * IMGW + sw;
        int rr = (hsf < IMGH - WSZ) ? 0 : ((hsf < IMGH - SSH) ? 1 : 2);
        int cc = (wsf < IMGW - WSZ) ? 0 : ((wsf < IMGW - SSH) ? 1 : 2);
        rid[tid] = rr * 3 + cc;
    }
    __syncthreads();

    // phase 1: gather window into LDS (bf16)
#if HAVE_ASYNC_LDS
    {
        float* stage = (float*)(smem + SM_QKV);   // f32 staging [64][384] (96 x 16B per row)
        for (int i = tid; i < NTOK * 96; i += 256) {
            int t = i / 96, c16 = i - t * 96;
            async_g2l_16B(stage + t * CDIM + c16 * 4,
                          x + (size_t)gidx[t] * CDIM + c16 * 4);
        }
        async_wait0();
        __syncthreads();
        for (int i = tid; i < NTOK * CDIM; i += 256) {
            int t = i / CDIM, c = i - t * CDIM;
            xw[t * 392 + c] = f2bf(stage[t * CDIM + c]);
        }
    }
#else
    for (int i = tid; i < NTOK * CDIM; i += 256) {
        int t = i / CDIM, c = i - t * CDIM;
        xw[t * 392 + c] = f2bf(x[(size_t)gidx[t] * CDIM + c]);
    }
#endif
    __syncthreads();

    // phase 2: QKV GEMM  [64,384] x [384,1152]^T  (288 tiles of 16x16)
    for (int tile = wave; tile < 288; tile += 8) {
        int mt = tile & 3, nt = tile >> 2;
        const unsigned short* arow = xw + ((mt << 4) + ln) * 392;
        const unsigned short* brow = qkvw + ((size_t)((nt << 4) + ln)) * CDIM;
        v8f acc = {0.f, 0.f, 0.f, 0.f, 0.f, 0.f, 0.f, 0.f};
#pragma unroll 4
        for (int ks = 0; ks < 12; ++ks) {
            __builtin_prefetch(brow + ks * 32 + 64, 0, 1);
            v16bf a = load_frag(arow + ks * 32, hf);
            v16bf bb = load_frag(brow + ks * 32, hf);
            acc = wmma_bf16(a, bb, acc);
        }
        int o = (nt << 4) + ln;
        float bias = (o < CDIM) ? qb[o] : ((o < 2 * CDIM) ? 0.f : vb[o - 2 * CDIM]);
#pragma unroll
        for (int r = 0; r < 8; ++r) {
            int m = (mt << 4) + r + (hf << 3);
            qkv[m * 1160 + o] = f2bf(acc[r] + bias);
        }
    }
    __syncthreads();

    // phase 3: per-head cosine attention
    for (int h = 0; h < NHEADS; ++h) {
        // 3a: normalize q,k rows (cosine attention)
        if (tid < 128) {
            int which = tid >> 6, t = tid & 63;
            const unsigned short* src = qkv + t * 1160 + which * CDIM + h * HDIM;
            float s = 0.f;
#pragma unroll
            for (int d = 0; d < HDIM; ++d) { float v = bf2f(src[d]); s += v * v; }
            float inv = 1.f / fmaxf(sqrtf(s), 1e-12f);
            unsigned short* dst = (which ? kn : qn) + t * 36;
#pragma unroll
            for (int d = 0; d < HDIM; ++d) dst[d] = f2bf(bf2f(src[d]) * inv);
        }
        __syncthreads();

        // 3b: S = qn @ kn^T * scale + 16*sigmoid(rpb) + shift-mask
        float sch = scale[h];
        const float* rpbh = rpb + h * (NTOK * NTOK);
        for (int tile = wave; tile < 16; tile += 8) {
            int mt = tile & 3, nt = tile >> 2;
            v16bf a = load_frag(qn + ((mt << 4) + ln) * 36, hf);
            v16bf bb = load_frag(kn + ((nt << 4) + ln) * 36, hf);
            v8f acc = {0.f, 0.f, 0.f, 0.f, 0.f, 0.f, 0.f, 0.f};
            acc = wmma_bf16(a, bb, acc);
            int n = (nt << 4) + ln;
#pragma unroll
            for (int r = 0; r < 8; ++r) {
                int m = (mt << 4) + r + (hf << 3);
                float v = acc[r] * sch + rpbh[m * NTOK + n]
                        + ((rid[m] != rid[n]) ? -100.f : 0.f);
                sc[m * 68 + n] = v;
            }
        }
        __syncthreads();

        // 3c: row softmax -> bf16 probabilities in pbuf
        if (tid < NTOK) {
            float mx = -1e30f;
            for (int j = 0; j < NTOK; ++j) mx = fmaxf(mx, sc[tid * 68 + j]);
            float s = 0.f;
            for (int j = 0; j < NTOK; ++j) {
                float e = expf(sc[tid * 68 + j] - mx);
                sc[tid * 68 + j] = e; s += e;
            }
            float inv = 1.f / s;
            for (int j = 0; j < NTOK; ++j) pbuf[tid * 68 + j] = f2bf(sc[tid * 68 + j] * inv);
        }
        __syncthreads();

        // 3d: out = P @ V  (8 tiles -> 1 per wave)
        {
            int mt = wave & 3, nt = wave >> 2;
            v8f acc = {0.f, 0.f, 0.f, 0.f, 0.f, 0.f, 0.f, 0.f};
#pragma unroll
            for (int ks = 0; ks < 2; ++ks) {
                v16bf a = load_frag(pbuf + ((mt << 4) + ln) * 68 + ks * 32, hf);
                v16bf bb = load_frag_strided(
                    qkv + (ks * 32) * 1160 + 2 * CDIM + h * HDIM + (nt << 4) + ln, 1160, hf);
                acc = wmma_bf16(a, bb, acc);
            }
            int d = (nt << 4) + ln;
#pragma unroll
            for (int r = 0; r < 8; ++r) {
                int m = (mt << 4) + r + (hf << 3);
                xw[m * 392 + h * HDIM + d] = f2bf(acc[r]);   // attn-out reuses xw region
            }
        }
        __syncthreads();
    }

    // phase 4: proj GEMM [64,384] x [384,384]^T  -> f32 into qkv region
    float* projf = (float*)(smem + SM_QKV);                   // stride 388
    for (int tile = wave; tile < 96; tile += 8) {
        int mt = tile & 3, nt = tile >> 2;
        const unsigned short* arow = xw + ((mt << 4) + ln) * 392;
        const unsigned short* brow = projw + ((size_t)((nt << 4) + ln)) * CDIM;
        v8f acc = {0.f, 0.f, 0.f, 0.f, 0.f, 0.f, 0.f, 0.f};
#pragma unroll 4
        for (int ks = 0; ks < 12; ++ks) {
            __builtin_prefetch(brow + ks * 32 + 64, 0, 1);
            v16bf a = load_frag(arow + ks * 32, hf);
            v16bf bb = load_frag(brow + ks * 32, hf);
            acc = wmma_bf16(a, bb, acc);
        }
        int n = (nt << 4) + ln;
        float pb = projb[n];
#pragma unroll
        for (int r = 0; r < 8; ++r) {
            int m = (mt << 4) + r + (hf << 3);
            projf[m * 388 + n] = acc[r] + pb;
        }
    }
    __syncthreads();

    // phase 5: LayerNorm(post-norm) + residual, scattered store of x1
    if (tid < NTOK) {
        const float* row = projf + tid * 388;
        float s = 0.f, s2 = 0.f;
        for (int c = 0; c < CDIM; ++c) { float v = row[c]; s += v; s2 += v * v; }
        float mu = s * (1.f / CDIM);
        float var = s2 * (1.f / CDIM) - mu * mu;
        float rinv = rsqrtf(var + 1e-5f);
        size_t g = (size_t)gidx[tid] * CDIM;
        for (int c = 0; c < CDIM; ++c) {
            float v = (row[c] - mu) * rinv * g1[c] + b1[c] + x[g + c];
            x1out[g + c] = v;
        }
    }
}

// ---------------- fused MLP kernel ----------------
#define SM2_X 0                  // x1 bf16 [64][392] = 50176
#define SM2_H 50432              // hidden bf16 [64][1544] = 197632 ; f32 staging / y f32
#define SMEM_MLP (SM2_H + 197632)

__global__ __launch_bounds__(256)
void swin_mlp_kernel(const float* __restrict__ x1,
                     const unsigned short* __restrict__ w1, const float* __restrict__ bb1,
                     const unsigned short* __restrict__ w2, const float* __restrict__ bb2,
                     const float* __restrict__ g2, const float* __restrict__ b2v,
                     float* __restrict__ out) {
    extern __shared__ char smem[];
    unsigned short* x1b = (unsigned short*)(smem + SM2_X);   // stride 392
    unsigned short* hb  = (unsigned short*)(smem + SM2_H);   // stride 1544

    const int tid = threadIdx.x;
    const int wave = tid >> 5, lane = tid & 31;
    const int hf = lane >> 4, ln = lane & 15;
    const size_t t0 = (size_t)blockIdx.x * NTOK;

    // phase 1: load x1 rows -> bf16 LDS
#if HAVE_ASYNC_LDS
    {
        float* stage = (float*)(smem + SM2_H);   // f32 staging [64][384]
        const float* base = x1 + t0 * CDIM;
        for (int i = tid; i < NTOK * 96; i += 256) {
            async_g2l_16B(stage + i * 4, base + i * 4);   // rows are contiguous here
        }
        async_wait0();
        __syncthreads();
        for (int i = tid; i < NTOK * CDIM; i += 256) {
            int t = i / CDIM, c = i - t * CDIM;
            x1b[t * 392 + c] = f2bf(stage[i]);
        }
    }
#else
    for (int i = tid; i < NTOK * CDIM; i += 256) {
        int t = i / CDIM, c = i - t * CDIM;
        x1b[t * 392 + c] = f2bf(x1[(t0 + t) * CDIM + c]);
    }
#endif
    __syncthreads();

    // phase 2: H = gelu(x1 @ w1^T + b1)   [64,384]x[384,1536] -> 384 tiles
    for (int tile = wave; tile < 384; tile += 8) {
        int mt = tile & 3, nt = tile >> 2;
        const unsigned short* arow = x1b + ((mt << 4) + ln) * 392;
        const unsigned short* brow = w1 + ((size_t)((nt << 4) + ln)) * CDIM;
        v8f acc = {0.f, 0.f, 0.f, 0.f, 0.f, 0.f, 0.f, 0.f};
#pragma unroll 4
        for (int ks = 0; ks < 12; ++ks) {
            __builtin_prefetch(brow + ks * 32 + 64, 0, 1);
            v16bf a = load_frag(arow + ks * 32, hf);
            v16bf bb = load_frag(brow + ks * 32, hf);
            acc = wmma_bf16(a, bb, acc);
        }
        int o = (nt << 4) + ln;
        float bo = bb1[o];
#pragma unroll
        for (int r = 0; r < 8; ++r) {
            int m = (mt << 4) + r + (hf << 3);
            hb[m * 1544 + o] = f2bf(gelu_exact(acc[r] + bo));
        }
    }
    __syncthreads();

    // phase 3: y = H @ w2^T  [64,1536]x[1536,384] -> 96 tiles, 12/wave in registers
    v8f accs[12];
#pragma unroll
    for (int tt = 0; tt < 12; ++tt) {
        int tile = wave + tt * 8;
        int mt = tile & 3, nt = tile >> 2;
        const unsigned short* arow = hb + ((mt << 4) + ln) * 1544;
        const unsigned short* brow = w2 + ((size_t)((nt << 4) + ln)) * HID;
        v8f acc = {0.f, 0.f, 0.f, 0.f, 0.f, 0.f, 0.f, 0.f};
        for (int ks = 0; ks < 48; ++ks) {
            __builtin_prefetch(brow + ks * 32 + 64, 0, 1);
            v16bf a = load_frag(arow + ks * 32, hf);
            v16bf bb = load_frag(brow + ks * 32, hf);
            acc = wmma_bf16(a, bb, acc);
        }
        accs[tt] = acc;
    }
    __syncthreads();                    // all waves done reading hb
    float* yb = (float*)(smem + SM2_H); // stride 388, overwrites hb
#pragma unroll
    for (int tt = 0; tt < 12; ++tt) {
        int tile = wave + tt * 8;
        int mt = tile & 3, nt = tile >> 2;
        int n = (nt << 4) + ln;
        float bo = bb2[n];
#pragma unroll
        for (int r = 0; r < 8; ++r) {
            int m = (mt << 4) + r + (hf << 3);
            yb[m * 388 + n] = accs[tt][r] + bo;
        }
    }
    __syncthreads();

    // phase 4: LN2 + residual -> output
    if (tid < NTOK) {
        const float* row = yb + tid * 388;
        float s = 0.f, s2 = 0.f;
        for (int c = 0; c < CDIM; ++c) { float v = row[c]; s += v; s2 += v * v; }
        float mu = s * (1.f / CDIM);
        float var = s2 * (1.f / CDIM) - mu * mu;
        float rinv = rsqrtf(var + 1e-5f);
        size_t g = (t0 + tid) * CDIM;
        for (int c = 0; c < CDIM; ++c) {
            out[g + c] = x1[g + c] + (row[c] - mu) * rinv * g2[c] + b2v[c];
        }
    }
}

// ---------------- host launch ----------------
extern "C" void kernel_launch(void* const* d_in, const int* in_sizes, int n_in,
                              void* d_out, int out_size, void* d_ws, size_t ws_size,
                              hipStream_t stream) {
    const float* x       = (const float*)d_in[0];
    const float* qkv_w   = (const float*)d_in[1];
    const float* q_bias  = (const float*)d_in[2];
    const float* v_bias  = (const float*)d_in[3];
    const float* lscale  = (const float*)d_in[4];
    const float* cpb_w1  = (const float*)d_in[5];
    const float* cpb_b1  = (const float*)d_in[6];
    const float* cpb_w2  = (const float*)d_in[7];
    const float* proj_w  = (const float*)d_in[8];
    const float* proj_b  = (const float*)d_in[9];
    const float* ln1_g   = (const float*)d_in[10];
    const float* ln1_b   = (const float*)d_in[11];
    const float* ln2_g   = (const float*)d_in[12];
    const float* ln2_b   = (const float*)d_in[13];
    const float* mlp_w1  = (const float*)d_in[14];
    const float* mlp_b1  = (const float*)d_in[15];
    const float* mlp_w2  = (const float*)d_in[16];
    const float* mlp_b2  = (const float*)d_in[17];
    float* out = (float*)d_out;

    char* ws = (char*)d_ws;
    // workspace layout
    const size_t QKVW_OFF  = 0;                         // 1152*384 bf16
    const size_t PROJW_OFF = QKVW_OFF  + 884736;        // 384*384 bf16
    const size_t W1_OFF    = PROJW_OFF + 294912;        // 1536*384 bf16
    const size_t W2_OFF    = W1_OFF    + 1179648;       // 384*1536 bf16
    const size_t SCALE_OFF = W2_OFF    + 1179648;       // 12 f32 (padded)
    const size_t RPB_OFF   = SCALE_OFF + 256;           // 12*64*64 f32
    const size_t X1_OFF    = RPB_OFF   + 196608;        // 131072*384 f32

    unsigned short* qkvw_bf = (unsigned short*)(ws + QKVW_OFF);
    unsigned short* projw_bf = (unsigned short*)(ws + PROJW_OFF);
    unsigned short* w1_bf = (unsigned short*)(ws + W1_OFF);
    unsigned short* w2_bf = (unsigned short*)(ws + W2_OFF);
    float* scale = (float*)(ws + SCALE_OFF);
    float* rpb   = (float*)(ws + RPB_OFF);
    float* x1    = (float*)(ws + X1_OFF);

    // prep: weight conversion + CPB table
    cvt_bf16_kernel<<<(1152 * 384 + 255) / 256, 256, 0, stream>>>(qkv_w, qkvw_bf, 1152 * 384);
    cvt_bf16_kernel<<<(384 * 384 + 255) / 256, 256, 0, stream>>>(proj_w, projw_bf, 384 * 384);
    cvt_bf16_kernel<<<(1536 * 384 + 255) / 256, 256, 0, stream>>>(mlp_w1, w1_bf, 1536 * 384);
    cvt_bf16_kernel<<<(384 * 1536 + 255) / 256, 256, 0, stream>>>(mlp_w2, w2_bf, 384 * 1536);
    cpb_kernel<<<1, 256, 0, stream>>>(cpb_w1, cpb_b1, cpb_w2, lscale, scale, rpb);

    // fused attention + proj + LN1 residual
    swin_attn_kernel<<<NWIN, 256, SMEM_ATTN, stream>>>(
        x, qkvw_bf, q_bias, v_bias, scale, rpb, projw_bf, proj_b, ln1_g, ln1_b, x1);

    // fused MLP + LN2 residual
    swin_mlp_kernel<<<NWIN, 256, SMEM_MLP, stream>>>(
        x1, w1_bf, mlp_b1, w2_bf, mlp_b2, ln2_g, ln2_b, out);
}
